// AdaptedEditFlowsTransformer_80762565034119
// MI455X (gfx1250) — compile-verified, matrix operand order checked by reading, not run on hardware
//
#include <hip/hip_runtime.h>
#include <hip/hip_bf16.h>
#include <math.h>

// ---------------- problem constants (match reference) ----------------
constexpr int B  = 2;
constexpr int S  = 512;
constexpr int C  = 512;
constexpr int L  = 1024;
constexpr int H  = 1024;
constexpr int NH = 16;
constexpr int HD = 64;
constexpr int F  = 4096;
constexpr int NL = 4;
constexpr int V  = 32000;
constexpr int TD = 512;
constexpr float NEGB = -1.0e9f;

// ---------------- WMMA types ----------------
typedef __attribute__((ext_vector_type(16))) __bf16 v16bf;
typedef __attribute__((ext_vector_type(8)))  float  v8f;

union Frag16 { uint4 q[2]; v16bf v; };

__device__ __forceinline__ unsigned short f2bf(float f) {
  union { float f; unsigned u; } a; a.f = f;
  unsigned r = a.u + 0x7FFFu + ((a.u >> 16) & 1u); // round-to-nearest-even
  return (unsigned short)(r >> 16);
}

// =====================================================================
// Generic GEMM: Cout[M,N] = epilogue(A[M,K] @ W[K,N])
// mode: 0 = none, 1 = +res, 2 = +bias then SiLU, 3 = +bias
// A, W fp32 in global; converted to bf16 while staging into LDS.
// Block tile 128x128x32, 256 threads (8 wave32), wave tile 32x64.
// Double-buffered LDS (ping-pong): one barrier per k-step. Staging is
// two-phase: batch all 8 global_load_b128 (one loadcnt wait), then
// convert + ds_store, so load latency pipelines under WMMA work.
// =====================================================================
constexpr int BM = 128, BN = 128, BK = 32, LDT = 40; // LDT: padded bf16 k-stride
constexpr int TSZ = BM * LDT;                        // one tile's ushort count

__global__ __launch_bounds__(256) void gemm_wmma(
    const float* __restrict__ A, const float* __restrict__ W,
    const float* __restrict__ bias, const float* __restrict__ res,
    float* __restrict__ Cout, int M, int N, int K, int mode)
{
  __shared__ unsigned short As[2 * TSZ];
  __shared__ unsigned short Bs[2 * TSZ]; // holds W^T tile: [n][k]

  const int tid  = threadIdx.x;
  const int lane = tid & 31;
  const int wv   = tid >> 5;
  const int m0   = blockIdx.y * BM;
  const int n0   = blockIdx.x * BN;
  const int mbase = (wv & 3) * 32;   // 4 waves along M
  const int nbase = (wv >> 2) * 64;  // 2 waves along N
  const int frow = lane & 15;        // row-in-fragment
  const int kb   = (lane >> 4) << 3; // k base (0 or 8) per ISA 16-bit A layout

  // per-thread staging coordinates (constant across k-steps)
  const int ar = tid >> 1;                 // A row 0..127   (8 float4/row/thread pair)
  const int ac = (tid & 1) << 4;           // A col 0 or 16  (covers 4 float4 each)
  const int wr = tid >> 3;                 // W k-row 0..31
  const int wc = (tid & 7) << 4;           // W n-col, 4 float4 each

  v8f acc[2][4];
#pragma unroll
  for (int i = 0; i < 2; ++i)
#pragma unroll
    for (int j = 0; j < 4; ++j)
#pragma unroll
      for (int e = 0; e < 8; ++e) acc[i][j][e] = 0.0f;

  // Stage one 128x32 (A) + 32x128 (W, transposed) tile pair into buffer pb.
  // Phase 1: issue all global loads (kept in flight together).
  // Phase 2: convert to bf16 and store to LDS.
  auto stage_tiles = [&](int kk, int pb) {
    float4 av[4], wvv[4];
#pragma unroll
    for (int i = 0; i < 4; ++i) {
      const int f = tid + i * 256;          // 0..1023 float4 slots
      const int r = f >> 3;                 // A row 0..127
      const int c = (f & 7) << 2;           // A k col 0..28
      av[i] = *(const float4*)(A + (size_t)(m0 + r) * K + (kk + c));
    }
#pragma unroll
    for (int i = 0; i < 4; ++i) {
      const int f = tid + i * 256;
      const int r = f >> 5;                 // W k row 0..31
      const int c = (f & 31) << 2;          // W n col 0..124
      wvv[i] = *(const float4*)(W + (size_t)(kk + r) * N + (n0 + c));
    }
    unsigned short* as = &As[pb * TSZ];
    unsigned short* bs = &Bs[pb * TSZ];
#pragma unroll
    for (int i = 0; i < 4; ++i) {
      const int f = tid + i * 256;
      const int r = f >> 3;
      const int c = (f & 7) << 2;
      ushort4 u;
      u.x = f2bf(av[i].x); u.y = f2bf(av[i].y);
      u.z = f2bf(av[i].z); u.w = f2bf(av[i].w);
      *(ushort4*)(&as[r * LDT + c]) = u;
    }
#pragma unroll
    for (int i = 0; i < 4; ++i) {
      const int f = tid + i * 256;
      const int r = f >> 5;
      const int c = (f & 31) << 2;
      bs[(c + 0) * LDT + r] = f2bf(wvv[i].x);
      bs[(c + 1) * LDT + r] = f2bf(wvv[i].y);
      bs[(c + 2) * LDT + r] = f2bf(wvv[i].z);
      bs[(c + 3) * LDT + r] = f2bf(wvv[i].w);
    }
  };

  stage_tiles(0, 0);
  __syncthreads();

  int p = 0;
  for (int kk = 0; kk < K; kk += BK) {
    // prefetch global for the tile after next (double buffer covers kk+BK)
    if (kk + 2 * BK < K) {
      __builtin_prefetch(A + (size_t)(m0 + ar) * K + (kk + 2 * BK) + (ac & 31), 0, 1);
      __builtin_prefetch(W + (size_t)(kk + 2 * BK + wr) * N + n0 + wc, 0, 1);
    }
    // overlap: stage next tile pair into the other buffer
    if (kk + BK < K) stage_tiles(kk + BK, p ^ 1);

    // compute on current buffer
    const unsigned short* as = &As[p * TSZ];
    const unsigned short* bs = &Bs[p * TSZ];
    Frag16 afr[2], bfr[4];
#pragma unroll
    for (int fm = 0; fm < 2; ++fm) {
      const unsigned short* q = &as[(mbase + fm * 16 + frow) * LDT + kb];
      afr[fm].q[0] = *(const uint4*)(q);
      afr[fm].q[1] = *(const uint4*)(q + 16);
    }
#pragma unroll
    for (int fn = 0; fn < 4; ++fn) {
      const unsigned short* q = &bs[(nbase + fn * 16 + frow) * LDT + kb];
      bfr[fn].q[0] = *(const uint4*)(q);
      bfr[fn].q[1] = *(const uint4*)(q + 16);
    }
#pragma unroll
    for (int fm = 0; fm < 2; ++fm)
#pragma unroll
      for (int fn = 0; fn < 4; ++fn)
        acc[fm][fn] = __builtin_amdgcn_wmma_f32_16x16x32_bf16(
            false, afr[fm].v, false, bfr[fn].v, (short)0, acc[fm][fn],
            false, false);

    __syncthreads();
    p ^= 1;
  }

  // ---- epilogue: C/D layout -> lanes 0-15 M=r, lanes 16-31 M=r+8 ----
  const int lr = (lane >> 4) << 3;
  const int nc = lane & 15;
#pragma unroll
  for (int fm = 0; fm < 2; ++fm)
#pragma unroll
    for (int fn = 0; fn < 4; ++fn)
#pragma unroll
      for (int r = 0; r < 8; ++r) {
        const int m = m0 + mbase + fm * 16 + r + lr;
        const int n = n0 + nbase + fn * 16 + nc;
        float vv = acc[fm][fn][r];
        if (mode == 2 || mode == 3) vv += bias[n];
        if (mode == 2) vv = vv / (1.0f + __expf(-vv));
        if (mode == 1) vv += res[(size_t)m * N + n];
        Cout[(size_t)m * N + n] = vv;
      }
}

// =====================================================================
// Embedding gather + pad flags (pad iff token == pad_token)
// =====================================================================
__global__ __launch_bounds__(256) void embed_k(
    const int* __restrict__ tokens, const int* __restrict__ ctx,
    const int* __restrict__ padtok, const float* __restrict__ emb,
    float* __restrict__ x, int* __restrict__ padflag)
{
  const int idx = blockIdx.x * 256 + threadIdx.x; // B*L*H
  const int d   = idx % H;
  const int rr  = idx / H;
  const int pos = rr % L;
  const int b   = rr / L;
  const int tok = (pos < C) ? ctx[b * C + pos] : tokens[b * S + (pos - C)];
  x[idx] = emb[(size_t)tok * H + d];
  if (d == 0) padflag[b * L + pos] = (tok == padtok[0]) ? 1 : 0;
}

// =====================================================================
// RMSNorm: one block per row of H
// =====================================================================
__global__ __launch_bounds__(256) void rmsnorm_k(
    const float* __restrict__ x, const float* __restrict__ g,
    float* __restrict__ o)
{
  const int row = blockIdx.x; // B*L
  const float* xr = x + (size_t)row * H;
  float ss = 0.f;
  for (int d = threadIdx.x; d < H; d += 256) { const float v = xr[d]; ss += v * v; }
  __shared__ float red[256];
  red[threadIdx.x] = ss; __syncthreads();
  for (int off = 128; off; off >>= 1) {
    if (threadIdx.x < off) red[threadIdx.x] += red[threadIdx.x + off];
    __syncthreads();
  }
  const float inv = rsqrtf(red[0] / (float)H + 1e-6f);
  for (int d = threadIdx.x; d < H; d += 256)
    o[(size_t)row * H + d] = xr[d] * inv * g[d];
}

// =====================================================================
// RoPE on q and k (per (b,pos,h,d<32) rotate pair (d, d+32))
// =====================================================================
__global__ __launch_bounds__(256) void rope_qk(float* __restrict__ q,
                                               float* __restrict__ k)
{
  const int idx = blockIdx.x * 256 + threadIdx.x; // B*L*NH*32
  const int d   = idx & 31;
  const int h   = (idx >> 5) & (NH - 1);
  const int pos = (idx >> 9) & (L - 1);
  const int b   = idx >> 19;
  const float lg = 9.2103403719761836f; // ln(10000)
  const float freq = __expf(-lg * (float)d / 32.f);
  const float ang = (float)pos * freq;
  const float cv = __cosf(ang), sv = __sinf(ang);
  const size_t base = ((size_t)(b * L + pos)) * H + h * HD;
  const float q0 = q[base + d], q1 = q[base + d + 32];
  q[base + d]      = q0 * cv - q1 * sv;
  q[base + d + 32] = q1 * cv + q0 * sv;
  const float k0 = k[base + d], k1 = k[base + d + 32];
  k[base + d]      = k0 * cv - k1 * sv;
  k[base + d + 32] = k1 * cv + k0 * sv;
}

// =====================================================================
// Masked softmax attention: block = (8 q-rows) x (b,h); wave per row.
// Scores staged in LDS (8 x L fp32), two-pass stable softmax, P@V.
// Only ~3% of total FLOPs -> VALU is fine at the roofline.
// =====================================================================
__global__ __launch_bounds__(256) void attn_k(
    const float* __restrict__ Q, const float* __restrict__ Kt,
    const float* __restrict__ Vt, const float* __restrict__ randu,
    const int* __restrict__ padflag, float* __restrict__ Out)
{
  __shared__ float sq[8][HD];
  __shared__ float ss[8][L];
  const int tid = threadIdx.x, lane = tid & 31, w = tid >> 5;
  const int rb = blockIdx.x, h = blockIdx.y, b = blockIdx.z;

  for (int t = tid; t < 8 * HD; t += 256) {
    const int r = t / HD, d = t % HD;
    sq[r][d] = Q[((size_t)(b * L + rb * 8 + r)) * H + h * HD + d];
  }
  __syncthreads();

  const int row  = rb * 8 + w;
  const int qpad = padflag[b * L + row];
  float lmax = -3.4e38f;
  for (int j = lane; j < L; j += 32) {
    const float* kp = Kt + ((size_t)(b * L + j)) * H + h * HD;
    float dot = 0.f;
#pragma unroll
    for (int d = 0; d < HD; d += 4) {
      const float4 kv = *(const float4*)(kp + d);
      dot += sq[w][d] * kv.x + sq[w][d + 1] * kv.y +
             sq[w][d + 2] * kv.z + sq[w][d + 3] * kv.w;
    }
    const bool causal = (j <= row);
    const bool rnd = randu[(size_t)b * L * L + (size_t)row * L + j] < 0.1f;
    const bool ok  = !(qpad || padflag[b * L + j]);
    const float sv = dot * 0.125f + (((causal || rnd) && ok) ? 0.f : NEGB);
    ss[w][j] = sv;
    lmax = fmaxf(lmax, sv);
  }
#pragma unroll
  for (int o = 16; o; o >>= 1) lmax = fmaxf(lmax, __shfl_xor(lmax, o, 32));
  float lsum = 0.f;
  for (int j = lane; j < L; j += 32) {
    const float e = __expf(ss[w][j] - lmax);
    ss[w][j] = e;
    lsum += e;
  }
#pragma unroll
  for (int o = 16; o; o >>= 1) lsum += __shfl_xor(lsum, o, 32);
  const float inv = 1.f / lsum;
  __syncthreads();

  const int d0 = lane * 2;
  float a0 = 0.f, a1 = 0.f;
  for (int j = 0; j < L; ++j) {
    const float p = ss[w][j];
    const float* vr = Vt + ((size_t)(b * L + j)) * H + h * HD;
    a0 += p * vr[d0];
    a1 += p * vr[d0 + 1];
  }
  const size_t ob = ((size_t)(b * L + row)) * H + h * HD + d0;
  Out[ob]     = a0 * inv;
  Out[ob + 1] = a1 * inv;
}

// =====================================================================
// gate = silu(gate) * up   (elementwise over B*L*F)
// =====================================================================
__global__ __launch_bounds__(256) void silu_mul_k(float* __restrict__ g,
                                                  const float* __restrict__ u)
{
  const size_t i = (size_t)blockIdx.x * 256 + threadIdx.x;
  const float gv = g[i];
  g[i] = (gv / (1.f + __expf(-gv))) * u[i];
}

// =====================================================================
// xc = concat(hs, time-embedding)
// =====================================================================
__global__ __launch_bounds__(256) void build_xc_k(
    const float* __restrict__ nb, const float* __restrict__ t,
    float* __restrict__ xc)
{
  const int idx = blockIdx.x * 256 + threadIdx.x; // B*S*(H+TD)
  const int d = idx % (H + TD);
  const int r = idx / (H + TD);
  const int s = r % S, b = r / S;
  float val;
  if (d < H) {
    val = nb[((size_t)(b * L + C + s)) * H + d];
  } else {
    const int dd = d - H;
    const float tv = t[b];
    const float lg = 9.2103403719761836f; // ln(10000)
    const float hf = (float)(TD / 2);
    if (dd < TD / 2) val = __sinf(tv * __expf(-lg * (float)dd / hf));
    else             val = __cosf(tv * __expf(-lg * (float)(dd - TD / 2) / hf));
  }
  xc[idx] = val;
}

// =====================================================================
// rate head: out3 = softplus(hidden(512) @ w2(512x3) + b2) * mask
// =====================================================================
__global__ __launch_bounds__(128) void rate_head_k(
    const float* __restrict__ hh, const float* __restrict__ w2,
    const float* __restrict__ b2, const int* __restrict__ padflag,
    float* __restrict__ out)
{
  const int row = blockIdx.x; // B*S
  const int b = row / S, s = row % S;
  const float* hr = hh + (size_t)row * TD;
  float p0 = 0.f, p1 = 0.f, p2 = 0.f;
  for (int kq = threadIdx.x; kq < TD; kq += 128) {
    const float hv = hr[kq];
    p0 += hv * w2[kq * 3 + 0];
    p1 += hv * w2[kq * 3 + 1];
    p2 += hv * w2[kq * 3 + 2];
  }
  __shared__ float r0[128], r1[128], r2[128];
  r0[threadIdx.x] = p0; r1[threadIdx.x] = p1; r2[threadIdx.x] = p2;
  __syncthreads();
  for (int o = 64; o; o >>= 1) {
    if (threadIdx.x < o) {
      r0[threadIdx.x] += r0[threadIdx.x + o];
      r1[threadIdx.x] += r1[threadIdx.x + o];
      r2[threadIdx.x] += r2[threadIdx.x + o];
    }
    __syncthreads();
  }
  if (threadIdx.x == 0) {
    const float mm = padflag[b * L + C + s] ? 0.f : 1.f;
    const float vv[3] = { r0[0] + b2[0], r1[0] + b2[1], r2[0] + b2[2] };
#pragma unroll
    for (int o2 = 0; o2 < 3; ++o2) {
      const float vx = vv[o2];
      const float sp = fmaxf(vx, 0.f) + log1pf(__expf(-fabsf(vx)));
      out[(size_t)row * 3 + o2] = sp * mm;
    }
  }
}

// =====================================================================
// row softmax over V, then multiply by pad mask
// =====================================================================
__global__ __launch_bounds__(256) void softmax_vocab_k(
    float* __restrict__ logits, const int* __restrict__ padflag)
{
  const int row = blockIdx.x; // B*S
  const int b = row / S, s = row % S;
  float* p = logits + (size_t)row * V;
  __shared__ float red[256];
  float lm = -3.4e38f;
  for (int j = threadIdx.x; j < V; j += 256) lm = fmaxf(lm, p[j]);
  red[threadIdx.x] = lm; __syncthreads();
  for (int o = 128; o; o >>= 1) {
    if (threadIdx.x < o) red[threadIdx.x] = fmaxf(red[threadIdx.x], red[threadIdx.x + o]);
    __syncthreads();
  }
  const float m = red[0]; __syncthreads();
  float ls = 0.f;
  for (int j = threadIdx.x; j < V; j += 256) {
    const float e = __expf(p[j] - m);
    p[j] = e;
    ls += e;
  }
  red[threadIdx.x] = ls; __syncthreads();
  for (int o = 128; o; o >>= 1) {
    if (threadIdx.x < o) red[threadIdx.x] += red[threadIdx.x + o];
    __syncthreads();
  }
  const float scale = padflag[b * L + C + s] ? 0.f : (1.f / red[0]);
  for (int j = threadIdx.x; j < V; j += 256) p[j] *= scale;
}

// =====================================================================
// host-side orchestration
// =====================================================================
extern "C" void kernel_launch(void* const* d_in, const int* in_sizes, int n_in,
                              void* d_out, int out_size, void* d_ws, size_t ws_size,
                              hipStream_t stream) {
  (void)in_sizes; (void)n_in; (void)out_size; (void)ws_size;

  const int*   tokens   = (const int*)  d_in[0];
  const float* t_in     = (const float*)d_in[1];
  const int*   ctx      = (const int*)  d_in[3];
  const int*   padtok   = (const int*)  d_in[4];
  const float* randu    = (const float*)d_in[5];
  const float* emb      = (const float*)d_in[6];
  const float* Wq       = (const float*)d_in[7];
  const float* Wk       = (const float*)d_in[8];
  const float* Wv       = (const float*)d_in[9];
  const float* Wo       = (const float*)d_in[10];
  const float* Wg       = (const float*)d_in[11];
  const float* Wu       = (const float*)d_in[12];
  const float* Wd       = (const float*)d_in[13];
  const float* ln1      = (const float*)d_in[14];
  const float* ln2      = (const float*)d_in[15];
  const float* final_ln = (const float*)d_in[16];
  const float* rate_w1  = (const float*)d_in[17];
  const float* rate_b1  = (const float*)d_in[18];
  const float* rate_w2  = (const float*)d_in[19];
  const float* rate_b2  = (const float*)d_in[20];
  const float* ins_w1   = (const float*)d_in[21];
  const float* ins_b1   = (const float*)d_in[22];
  const float* ins_w2   = (const float*)d_in[23];
  const float* ins_b2   = (const float*)d_in[24];
  const float* sub_w1   = (const float*)d_in[25];
  const float* sub_b1   = (const float*)d_in[26];
  const float* sub_w2   = (const float*)d_in[27];
  const float* sub_b2   = (const float*)d_in[28];

  float* out = (float*)d_out;

  // workspace layout (fp32 elements)
  float* ws   = (float*)d_ws;
  float* x    = ws;
  float* nb   = x    + (size_t)B * L * H;
  float* q    = nb   + (size_t)B * L * H;
  float* k    = q    + (size_t)B * L * H;
  float* v    = k    + (size_t)B * L * H;
  float* att  = v    + (size_t)B * L * H;
  float* gate = att  + (size_t)B * L * H;
  float* up   = gate + (size_t)B * L * F;
  float* xc   = up   + (size_t)B * L * F;
  float* hh   = xc   + (size_t)B * S * (H + TD);
  int*   padflag = (int*)(hh + (size_t)B * S * H);

  const int M  = B * L;   // 2048
  const int Ms = B * S;   // 1024

  // 1. embedding + pad flags
  embed_k<<<(B * L * H) / 256, 256, 0, stream>>>(tokens, ctx, padtok, emb, x, padflag);

  // 2. transformer layers
  const dim3 gHH(H / BN, M / BM);     // N=H
  const dim3 gHF(F / BN, M / BM);     // N=F
  for (int l = 0; l < NL; ++l) {
    rmsnorm_k<<<M, 256, 0, stream>>>(x, ln1 + (size_t)l * H, nb);
    gemm_wmma<<<gHH, 256, 0, stream>>>(nb, Wq + (size_t)l * H * H, nullptr, nullptr, q, M, H, H, 0);
    gemm_wmma<<<gHH, 256, 0, stream>>>(nb, Wk + (size_t)l * H * H, nullptr, nullptr, k, M, H, H, 0);
    gemm_wmma<<<gHH, 256, 0, stream>>>(nb, Wv + (size_t)l * H * H, nullptr, nullptr, v, M, H, H, 0);
    rope_qk<<<(B * L * NH * 32) / 256, 256, 0, stream>>>(q, k);
    attn_k<<<dim3(L / 8, NH, B), 256, 0, stream>>>(q, k, v, randu, padflag, att);
    gemm_wmma<<<gHH, 256, 0, stream>>>(att, Wo + (size_t)l * H * H, nullptr, x, x, M, H, H, 1);
    rmsnorm_k<<<M, 256, 0, stream>>>(x, ln2 + (size_t)l * H, nb);
    gemm_wmma<<<gHF, 256, 0, stream>>>(nb, Wg + (size_t)l * H * F, nullptr, nullptr, gate, M, F, H, 0);
    gemm_wmma<<<gHF, 256, 0, stream>>>(nb, Wu + (size_t)l * H * F, nullptr, nullptr, up, M, F, H, 0);
    silu_mul_k<<<(B * L * F) / 256, 256, 0, stream>>>(gate, up);
    gemm_wmma<<<gHH, 256, 0, stream>>>(gate, Wd + (size_t)l * F * H, nullptr, x, x, M, H, F, 1);
  }

  // 3. final norm + xc
  rmsnorm_k<<<M, 256, 0, stream>>>(x, final_ln, nb);
  build_xc_k<<<(B * S * (H + TD)) / 256, 256, 0, stream>>>(nb, t_in, xc);

  // 4. rate head
  gemm_wmma<<<dim3(TD / BN, Ms / BM), 256, 0, stream>>>(xc, rate_w1, rate_b1, nullptr, hh, Ms, TD, H + TD, 2);
  rate_head_k<<<Ms, 128, 0, stream>>>(hh, rate_w2, rate_b2, padflag, out);

  // 5. ins head
  float* ins_out = out + (size_t)B * S * 3;
  gemm_wmma<<<dim3(H / BN, Ms / BM), 256, 0, stream>>>(xc, ins_w1, ins_b1, nullptr, hh, Ms, H, H + TD, 2);
  gemm_wmma<<<dim3(V / BN, Ms / BM), 256, 0, stream>>>(hh, ins_w2, ins_b2, nullptr, ins_out, Ms, V, H, 3);
  softmax_vocab_k<<<Ms, 256, 0, stream>>>(ins_out, padflag);

  // 6. sub head
  float* sub_out = ins_out + (size_t)B * S * V;
  gemm_wmma<<<dim3(H / BN, Ms / BM), 256, 0, stream>>>(xc, sub_w1, sub_b1, nullptr, hh, Ms, H, H + TD, 2);
  gemm_wmma<<<dim3(V / BN, Ms / BM), 256, 0, stream>>>(hh, sub_w2, sub_b2, nullptr, sub_out, Ms, V, H, 3);
  softmax_vocab_k<<<Ms, 256, 0, stream>>>(sub_out, padflag);
}
